// GraphStateEncoder_53652731462304
// MI455X (gfx1250) — matrix operand order repaired and checked by reading, not compiled, not run on hardware
//
#include <hip/hip_runtime.h>

typedef __bf16 bf16;
typedef __bf16 v16bf __attribute__((ext_vector_type(16)));
typedef float  v8f   __attribute__((ext_vector_type(8)));
typedef unsigned int v8u __attribute__((ext_vector_type(8)));
typedef unsigned int v4u __attribute__((ext_vector_type(4)));

union APack {
    v16bf v;
    v4u q[2];
};
union BPack {
    v16bf v;
    v8u w;
};

__device__ __forceinline__ v8f wmma_bf16(const v16bf& a, const v16bf& b, v8f c) {
    return __builtin_amdgcn_wmma_f32_16x16x32_bf16(false, a, false, b,
                                                   (short)0, c, false, false);
}

// Load the 16x32 bf16 A-fragment for this lane from LDS.
// base = &ldsU32[arow * (LDX/2) + hi*4]; fragment = dwords {ks*16+0..3, ks*16+8..11}
__device__ __forceinline__ APack load_a(const unsigned int* base, int dw_off) {
    APack a;
    a.q[0] = *(const v4u*)(base + dw_off);
    a.q[1] = *(const v4u*)(base + dw_off + 8);
    return a;
}

// -------------------------------------------------------------------------
// Pre-swizzle a K x 128 fp32 weight matrix into bf16 in the exact B-operand
// lane layout of v_wmma_f32_16x16x32_bf16 (16 contiguous bf16 per lane per
// (nt,ks) tile -> one 32B load in the GEMM kernels).
// -------------------------------------------------------------------------
__global__ void pack_weights(const float* __restrict__ W, bf16* __restrict__ P, int K) {
    const int ksteps = K >> 5;
    const int total = K * 128;
    for (int t = blockIdx.x * blockDim.x + threadIdx.x; t < total;
         t += gridDim.x * blockDim.x) {
        int j    = t & 15;
        int lane = (t >> 4) & 31;
        int rest = t >> 9;
        int ks   = rest % ksteps;
        int nt   = rest / ksteps;
        int vreg = j >> 1;
        int h    = j & 1;
        int k    = ks * 32 + ((lane >> 4) << 3) + (j & 7) + ((j >> 3) << 4);
        int col  = nt * 16 + (lane & 15);
        P[t] = (bf16)W[(size_t)k * 128 + col];
    }
}

// -------------------------------------------------------------------------
// Fused 2-layer MLP: OUT = relu(X @ W1 + b1) @ W2 + b2, output width 128.
// 256 threads = 8 waves; wave = (row-tile rt = w>>1, N-half nh = w&1).
// Each wave: 16 rows x 64 cols (4 N-tiles), acc = 32 VGPRs.
// -------------------------------------------------------------------------
template <int K1, bool BF16OUT>
__global__ __launch_bounds__(256) void mlp_kernel(
    const float* __restrict__ X,
    const bf16* __restrict__ W1p, const float* __restrict__ B1,
    const bf16* __restrict__ W2p, const float* __restrict__ B2,
    void* __restrict__ OUT, int rows_in, int rows_out)
{
    constexpr int KS1  = K1 / 32;
    constexpr int LDX  = K1 + 8;      // halves; multiple of 8 -> 16B-aligned rows
    constexpr int LDXU = LDX / 2;     // dwords per row
    constexpr int LDH  = 136;         // H stride (halves), 128+8
    constexpr int LDHU = 68;
    __shared__ __align__(16) bf16 ldsX[64 * LDX];
    __shared__ __align__(16) bf16 ldsH[4][16 * LDH];

    const int t    = threadIdx.x;
    const int lane = t & 31;
    const int wave = t >> 5;
    const int rt   = wave >> 1;       // row-tile 0..3
    const int nh   = wave & 1;        // N-half 0..1
    const int m    = lane & 15;
    const int hi   = lane >> 4;
    const int rowBlock = blockIdx.x * 64;

    // stage X (fp32 -> bf16) into LDS, coalesced
    for (int i = t; i < 64 * K1; i += 256) {
        int r = i / K1, c = i - r * K1;
        int row = rowBlock + r;
        float v = (row < rows_in) ? X[(size_t)row * K1 + c] : 0.0f;
        ldsX[r * LDX + c] = (bf16)v;
    }
    __syncthreads();

    const v8f zero = {0.f, 0.f, 0.f, 0.f, 0.f, 0.f, 0.f, 0.f};
    v8f acc[4];
#pragma unroll
    for (int i = 0; i < 4; ++i) acc[i] = zero;

    const int arow = rt * 16 + m;
    const unsigned int* Xbase = (const unsigned int*)ldsX + arow * LDXU + hi * 4;

    // ---- GEMM1: [16 x K1] @ [K1 x 64(half)] ----
#pragma unroll 1
    for (int ks = 0; ks < KS1; ++ks) {
        APack a = load_a(Xbase, ks * 16);
#pragma unroll
        for (int nt = 0; nt < 4; ++nt) {
            int ntg = nh * 4 + nt;
            BPack b;
            b.w = ((const v8u*)W1p)[(ntg * KS1 + ks) * 32 + lane];
            acc[nt] = wmma_bf16(a.v, b.v, acc[nt]);
        }
    }
    // bias + relu -> LDS H tile (bf16) for this row-tile / N-half
    bf16* H = &ldsH[rt][0];
#pragma unroll
    for (int nt = 0; nt < 4; ++nt) {
        int col = (nh * 4 + nt) * 16 + m;
        float bias = B1[col];
#pragma unroll
        for (int vi = 0; vi < 8; ++vi) {
            int r = vi + (hi << 3);
            float val = acc[nt][vi] + bias;
            H[r * LDH + col] = (bf16)(val > 0.f ? val : 0.f);
        }
    }
    __syncthreads();   // both N-halves of H must be complete (K=128 for GEMM2)

    // ---- GEMM2: [16 x 128] @ [128 x 64(half)] ----
#pragma unroll
    for (int i = 0; i < 4; ++i) acc[i] = zero;
    const unsigned int* Hbase = (const unsigned int*)H + m * LDHU + hi * 4;
#pragma unroll 1
    for (int ks = 0; ks < 4; ++ks) {
        APack a = load_a(Hbase, ks * 16);
#pragma unroll
        for (int nt = 0; nt < 4; ++nt) {
            int ntg = nh * 4 + nt;
            BPack b;
            b.w = ((const v8u*)W2p)[(ntg * 4 + ks) * 32 + lane];
            acc[nt] = wmma_bf16(a.v, b.v, acc[nt]);
        }
    }
    const int row0 = rowBlock + rt * 16;
#pragma unroll
    for (int nt = 0; nt < 4; ++nt) {
        int col = (nh * 4 + nt) * 16 + m;
        float bias = B2[col];
#pragma unroll
        for (int vi = 0; vi < 8; ++vi) {
            int row = row0 + vi + (hi << 3);
            if (row < rows_out) {
                float val = acc[nt][vi] + bias;
                if constexpr (BF16OUT)
                    ((bf16*)OUT)[(size_t)row * 128 + col] = (bf16)val;
                else
                    ((float*)OUT)[(size_t)row * 128 + col] = val;
            }
        }
    }
}

// -------------------------------------------------------------------------
// Message kernel: per edge, A = concat[h_src | edge_emb | h_dst] (K=384),
// fused 384->128(relu)->128 MLP for both directions from one staged LDS tile
// (bwd = per-ks segment shift), atomic f32 scatter into new_emb.
// 128 threads = 4 waves; wave = (edge-tile rt = w>>1, N-half nh = w&1).
// -------------------------------------------------------------------------
__global__ __launch_bounds__(128) void msg_kernel(
    const float* __restrict__ node_emb,
    const bf16* __restrict__ edge_emb,
    const int* __restrict__ edge_list,
    const bf16* __restrict__ W1p, const float* __restrict__ B1,
    const bf16* __restrict__ W2p, const float* __restrict__ B2,
    float* __restrict__ new_emb,
    int E, int max_nodes)
{
    constexpr int K1 = 384, KS1 = 12;
    constexpr int LDX = K1 + 8, LDXU = LDX / 2;   // 392 halves, 196 dwords
    constexpr int LDH = 136, LDHU = 68;
    __shared__ __align__(16) bf16 ldsX[32 * LDX];
    __shared__ __align__(16) bf16 ldsH[2][16 * LDH];
    __shared__ int s_src[32], s_dst[32], s_valid[32];

    const int t    = threadIdx.x;
    const int lane = t & 31;
    const int wave = t >> 5;
    const int rt   = wave >> 1;
    const int nh   = wave & 1;
    const int m    = lane & 15;
    const int hi   = lane >> 4;
    const int rowBlock = blockIdx.x * 32;

    if (t < 32) {
        int row = rowBlock + t;
        int s = 0, d = 0, val = 0;
        if (row < E) {
            s = edge_list[(size_t)row * 2 + 0];
            d = edge_list[(size_t)row * 2 + 1];
            val = (s >= 0 && s < max_nodes && d >= 0 && d < max_nodes) ? 1 : 0;
            if (!val) { s = 0; d = 0; }
        }
        s_src[t] = s; s_dst[t] = d; s_valid[t] = val;
    }
    __syncthreads();

    // stage [h_src | edge_emb | h_dst] per edge into LDS as bf16
    for (int i = t; i < 32 * K1; i += 128) {
        int r = i / K1, c = i - r * K1;
        int row = rowBlock + r;
        float v;
        if (c < 128)       v = node_emb[(size_t)s_src[r] * 128 + c];
        else if (c < 256)  v = (row < E) ? (float)edge_emb[(size_t)row * 128 + (c - 128)] : 0.0f;
        else               v = node_emb[(size_t)s_dst[r] * 128 + (c - 256)];
        ldsX[r * LDX + c] = (bf16)v;
    }
    __syncthreads();

    const v8f zero = {0.f, 0.f, 0.f, 0.f, 0.f, 0.f, 0.f, 0.f};
    const int arow = rt * 16 + m;
    const unsigned int* Xbase = (const unsigned int*)ldsX + arow * LDXU + hi * 4;
    bf16* H = &ldsH[rt][0];
    const unsigned int* Hbase = (const unsigned int*)H + m * LDHU + hi * 4;

    for (int dir = 0; dir < 2; ++dir) {
        v8f acc[4];
#pragma unroll
        for (int i = 0; i < 4; ++i) acc[i] = zero;
        // GEMM1: K=384. Each 32-wide ks-step lies inside one 128-wide segment
        // (segment = ks>>2); bwd swaps segments 0<->2 -> dword offset +-128.
#pragma unroll 1
        for (int ks = 0; ks < KS1; ++ks) {
            int dw = ks * 16;
            if (dir) {
                int seg = ks >> 2;
                dw += (seg == 0) ? 128 : ((seg == 2) ? -128 : 0);
            }
            APack a = load_a(Xbase, dw);
#pragma unroll
            for (int nt = 0; nt < 4; ++nt) {
                int ntg = nh * 4 + nt;
                BPack b;
                b.w = ((const v8u*)W1p)[(ntg * KS1 + ks) * 32 + lane];
                acc[nt] = wmma_bf16(a.v, b.v, acc[nt]);
            }
        }
        // bias + relu -> H
#pragma unroll
        for (int nt = 0; nt < 4; ++nt) {
            int col = (nh * 4 + nt) * 16 + m;
            float bias = B1[col];
#pragma unroll
            for (int vi = 0; vi < 8; ++vi) {
                int r = vi + (hi << 3);
                float val = acc[nt][vi] + bias;
                H[r * LDH + col] = (bf16)(val > 0.f ? val : 0.f);
            }
        }
        __syncthreads();   // full K=128 of H needed by both N-half waves

        // GEMM2: K=128
#pragma unroll
        for (int i = 0; i < 4; ++i) acc[i] = zero;
#pragma unroll 1
        for (int ks = 0; ks < 4; ++ks) {
            APack a = load_a(Hbase, ks * 16);
#pragma unroll
            for (int nt = 0; nt < 4; ++nt) {
                int ntg = nh * 4 + nt;
                BPack b;
                b.w = ((const v8u*)W2p)[(ntg * 4 + ks) * 32 + lane];
                acc[nt] = wmma_bf16(a.v, b.v, acc[nt]);
            }
        }
        // scatter: atomic f32 add into new_emb[target_node]
#pragma unroll
        for (int nt = 0; nt < 4; ++nt) {
            int col = (nh * 4 + nt) * 16 + m;
            float bias = B2[col];
#pragma unroll
            for (int vi = 0; vi < 8; ++vi) {
                int rl = rt * 16 + vi + (hi << 3);
                if (s_valid[rl]) {
                    int node = dir ? s_src[rl] : s_dst[rl];
                    float val = acc[nt][vi] + bias;
                    (void)__hip_atomic_fetch_add(&new_emb[(size_t)node * 128 + col], val,
                                                 __ATOMIC_RELAXED, __HIP_MEMORY_SCOPE_AGENT);
                }
            }
        }
        __syncthreads();   // protect H before dir=1 rewrites it
    }
}

// -------------------------------------------------------------------------
extern "C" void kernel_launch(void* const* d_in, const int* in_sizes, int n_in,
                              void* d_out, int out_size, void* d_ws, size_t ws_size,
                              hipStream_t stream)
{
    const float* node_features = (const float*)d_in[0];
    const int*   edge_list     = (const int*)d_in[1];
    const float* edge_features = (const float*)d_in[2];
    // d_in[3] = max_nodes (scalar, device) — value == out_size/128, derived on host
    const float* ne_w1 = (const float*)d_in[4];
    const float* ne_b1 = (const float*)d_in[5];
    const float* ne_w2 = (const float*)d_in[6];
    const float* ne_b2 = (const float*)d_in[7];
    const float* ee_w1 = (const float*)d_in[8];
    const float* ee_b1 = (const float*)d_in[9];
    const float* ee_w2 = (const float*)d_in[10];
    const float* ee_b2 = (const float*)d_in[11];
    const float* ml_w1 = (const float*)d_in[12];
    const float* ml_b1 = (const float*)d_in[13];
    const float* ml_w2 = (const float*)d_in[14];
    const float* ml_b2 = (const float*)d_in[15];
    const float* ag_w1 = (const float*)d_in[16];
    const float* ag_b1 = (const float*)d_in[17];
    const float* ag_w2 = (const float*)d_in[18];
    const float* ag_b2 = (const float*)d_in[19];

    const int n    = in_sizes[0] / 64;    // actual node count (NF=64)
    const int E    = in_sizes[1] / 2;     // edge count
    const int maxN = out_size / 128;      // == max_nodes (output rows)

    char* ws = (char*)d_ws;
    size_t off = 0;
    auto take = [&](size_t bytes) -> char* {
        char* p = ws + off;
        off = (off + bytes + 255) & ~(size_t)255;
        return p;
    };
    bf16* pw_ne1    = (bf16*)take((size_t)64  * 128 * 2);
    bf16* pw_ne2    = (bf16*)take((size_t)128 * 128 * 2);
    bf16* pw_ee1    = (bf16*)take((size_t)32  * 128 * 2);
    bf16* pw_ee2    = (bf16*)take((size_t)128 * 128 * 2);
    bf16* pw_ml1[2], *pw_ml2[2];
    for (int l = 0; l < 2; ++l) {
        pw_ml1[l] = (bf16*)take((size_t)384 * 128 * 2);
        pw_ml2[l] = (bf16*)take((size_t)128 * 128 * 2);
    }
    bf16* pw_ag1    = (bf16*)take((size_t)128 * 128 * 2);
    bf16* pw_ag2    = (bf16*)take((size_t)128 * 128 * 2);
    float* node_emb = (float*)take((size_t)maxN * 128 * 4);
    float* new_emb  = (float*)take((size_t)maxN * 128 * 4);
    bf16*  edge_emb = (bf16*)take((size_t)E * 128 * 2);

    auto pg = [](int K) { return dim3((unsigned)((K * 128 + 255) / 256)); };
    pack_weights<<<pg(64),  256, 0, stream>>>(ne_w1, pw_ne1, 64);
    pack_weights<<<pg(128), 256, 0, stream>>>(ne_w2, pw_ne2, 128);
    pack_weights<<<pg(32),  256, 0, stream>>>(ee_w1, pw_ee1, 32);
    pack_weights<<<pg(128), 256, 0, stream>>>(ee_w2, pw_ee2, 128);
    for (int l = 0; l < 2; ++l) {
        pack_weights<<<pg(384), 256, 0, stream>>>(ml_w1 + (size_t)l * 384 * 128, pw_ml1[l], 384);
        pack_weights<<<pg(128), 256, 0, stream>>>(ml_w2 + (size_t)l * 128 * 128, pw_ml2[l], 128);
    }
    pack_weights<<<pg(128), 256, 0, stream>>>(ag_w1, pw_ag1, 128);
    pack_weights<<<pg(128), 256, 0, stream>>>(ag_w2, pw_ag2, 128);

    const int gN = (maxN + 63) / 64;
    const int gE = (E + 63) / 64;
    // node encoder: 64 -> 128 -> 128 (rows beyond n are implicit zero pad)
    mlp_kernel<64, false><<<gN, 256, 0, stream>>>(
        node_features, pw_ne1, ne_b1, pw_ne2, ne_b2, node_emb, n, maxN);
    // edge encoder: 32 -> 128 -> 128, output bf16 (WMMA input for messages)
    mlp_kernel<32, true><<<gE, 256, 0, stream>>>(
        edge_features, pw_ee1, ee_b1, pw_ee2, ee_b2, edge_emb, E, E);

    const int gM = (E + 31) / 32;
    for (int l = 0; l < 2; ++l) {
        hipMemcpyAsync(new_emb, node_emb, (size_t)maxN * 128 * 4,
                       hipMemcpyDeviceToDevice, stream);
        msg_kernel<<<gM, 128, 0, stream>>>(
            node_emb, edge_emb, edge_list,
            pw_ml1[l], ml_b1 + (size_t)l * 128,
            pw_ml2[l], ml_b2 + (size_t)l * 128,
            new_emb, E, maxN);
        float* outp = (l == 1) ? (float*)d_out : node_emb;
        mlp_kernel<128, false><<<gN, 256, 0, stream>>>(
            new_emb, pw_ag1, ag_b1, pw_ag2, ag_b2, outp, maxN, maxN);
    }
    (void)n_in; (void)ws_size;
}
